// SCPositionwiseFeedForward_high_82317343195168
// MI455X (gfx1250) — matrix operand order repaired and checked by reading, not compiled, not run on hardware
//
#include <hip/hip_runtime.h>
#include <hip/hip_bf16.h>

// ---------------------------------------------------------------------------
// SCPositionwiseFeedForward for MI455X (gfx1250, wave32, WMMA)
//   h   = mish(w1 @ x^T + b1)                 (per-batch GEMM, f16 WMMA)
//   c   = depthwise_conv_K9(h, wnorm(d_w)) + d_b
//   a2  = mish(pwn^T @ c + p_b)               (per-batch GEMM, f16 WMMA)
//   out = (w2 @ a2 + b2)^T + x                (per-batch GEMM, f16 WMMA)
// ---------------------------------------------------------------------------

typedef _Float16 f16;
typedef __attribute__((ext_vector_type(16))) _Float16     v16h;
typedef __attribute__((ext_vector_type(8)))  float        v8f;
typedef __attribute__((ext_vector_type(4)))  unsigned int v4u;
typedef int v4i __attribute__((vector_size(16)));   // matches builtin param type

#define B_   16
#define T_   1024
#define D_   256
#define C_   1024
#define K_   9
#define PAD_ 4
#define EPSF 1e-12f

// GEMM tiling: 128x128 block tile, K-chunk 32, 256 threads = 8 waves (4x2)
#define BM 128
#define BN 128
#define BK 32
#define STRIDE_H 40   // f16 row stride in LDS: 80B = 16B-aligned, conflict-free
#define NTHREADS 256

#if defined(__has_builtin)
#if __has_builtin(__builtin_amdgcn_global_load_async_to_lds_b128) && \
    __has_builtin(__builtin_amdgcn_s_wait_asynccnt)
#define HAVE_ASYNC_LDS 1
#endif
#endif

__device__ __forceinline__ float mishf(float x) {
    float sp = (x > 20.f) ? x : log1pf(expf(x));
    return x * tanhf(sp);
}

union FragU { v16h h; unsigned int u[8]; };

// A fragment, 16x32 f16 (MxK): lane = (khalf<<4)|m ; VGPR v0..3: K=8*khalf+2v,
// v4..7: K=16+8*khalf+2(v-4).  Tile stored row-major [m][k], stride STRIDE_H.
__device__ __forceinline__ v16h frag_a(const f16* tile, int m, int lhi) {
    const unsigned int* r = (const unsigned int*)(tile + m * STRIDE_H);
    int kb = 4 * lhi;
    FragU f;
#pragma unroll
    for (int v = 0; v < 4; ++v) { f.u[v] = r[kb + v]; f.u[v + 4] = r[kb + v + 8]; }
    return f.h;
}

// B fragment, 32x16 f16 (KxN): lane = (khalf<<4)|n ; VGPR v0..3: K=8*khalf+2v,
// v4..7: K=16+8*khalf+2(v-4).
// Tile stored row-major [n][k] (i.e. B transposed), stride STRIDE_H.
__device__ __forceinline__ v16h frag_b(const f16* tile, int n, int lhi) {
    const unsigned int* r = (const unsigned int*)(tile + n * STRIDE_H);
    int kb = 4 * lhi;
    FragU f;
#pragma unroll
    for (int v = 0; v < 4; ++v) { f.u[v] = r[kb + v]; f.u[v + 4] = r[kb + v + 8]; }
    return f.h;
}

// Stage a 128-row x 32-k f16 tile whose source rows are contiguous in k.
// Each thread moves 2 x 16B.  Uses async global->LDS DMA when available.
__device__ __forceinline__ void stage_direct(f16* tile, const f16* src,
                                             int srcStride, int tid) {
#pragma unroll
    for (int i = 0; i < 2; ++i) {
        int ci  = tid + i * NTHREADS;
        int row = ci >> 2;
        int seg = ci & 3;
        const f16* g = src + (long)row * srcStride + seg * 8;
        f16*       l = tile + row * STRIDE_H + seg * 8;
#if defined(HAVE_ASYNC_LDS)
        __builtin_amdgcn_global_load_async_to_lds_b128(
            (__attribute__((address_space(1))) v4i*)(void*)g,
            (__attribute__((address_space(3))) v4i*)(void*)l, 0, 0);
#else
        *(v4u*)l = *(const v4u*)g;
#endif
    }
}

// Stage a tile whose source rows are contiguous in the *row* direction
// (k is the strided axis): tile[r][k] = src[k*srcStride + r].
__device__ __forceinline__ void stage_transposed(f16* tile, const f16* src,
                                                 long srcStride, int tid) {
    int r     = tid & 127;       // coalesced global reads along r
    int kbase = tid >> 7;        // 0..1
#pragma unroll
    for (int kk = 0; kk < BK; kk += 2) {
        int k = kk + kbase;
        tile[r * STRIDE_H + k] = src[(long)k * srcStride + r];
    }
}

__device__ __forceinline__ void stage_wait_barrier() {
#if defined(HAVE_ASYNC_LDS)
    __builtin_amdgcn_s_wait_asynccnt(0);
#endif
    __syncthreads();
}

// ---------------------------------------------------------------------------
// f32 -> f16 pack
// ---------------------------------------------------------------------------
__global__ void k_f32_to_f16(const float* __restrict__ src, f16* __restrict__ dst,
                             int n) {
    int i = blockIdx.x * blockDim.x + threadIdx.x;
    if (i < n) dst[i] = (f16)src[i];
}

// ---------------------------------------------------------------------------
// Depthwise weight norm: norm over channel axis per (b,k)
// dwn[b,c,k] = d_w[b,c,0,k] / max(||d_w[b,:,0,k]||, eps) * d_g[b,c]
// ---------------------------------------------------------------------------
__global__ __launch_bounds__(NTHREADS)
void k_dnorm(const float* __restrict__ d_w, const float* __restrict__ d_g,
             float* __restrict__ dwn) {
    int bk = blockIdx.x;
    int b = bk / K_, k = bk % K_;
    __shared__ float red[NTHREADS];
    float s = 0.f;
    for (int c = threadIdx.x; c < C_; c += NTHREADS) {
        float v = d_w[((long)b * C_ + c) * K_ + k];
        s += v * v;
    }
    red[threadIdx.x] = s;
    __syncthreads();
    for (int off = NTHREADS / 2; off > 0; off >>= 1) {
        if ((int)threadIdx.x < off) red[threadIdx.x] += red[threadIdx.x + off];
        __syncthreads();
    }
    float inv = 1.0f / fmaxf(sqrtf(red[0]), EPSF);
    for (int c = threadIdx.x; c < C_; c += NTHREADS) {
        long idx = ((long)b * C_ + c) * K_ + k;
        dwn[idx] = d_w[idx] * inv * d_g[b * C_ + c];
    }
}

// ---------------------------------------------------------------------------
// Pointwise weight norm: norm over input channel per (b,o)
// ---------------------------------------------------------------------------
__global__ __launch_bounds__(256)
void k_pnorm(const float* __restrict__ p_w, float* __restrict__ pnorm) {
    int b = blockIdx.y;
    int o = blockIdx.x * blockDim.x + threadIdx.x;
    float s = 0.f;
    for (int c = 0; c < C_; ++c) {
        float v = p_w[((long)b * C_ + c) * C_ + o];
        s += v * v;
    }
    pnorm[b * C_ + o] = fmaxf(sqrtf(s), EPSF);
}

__global__ __launch_bounds__(256)
void k_pwn(const float* __restrict__ p_w, const float* __restrict__ p_g,
           const float* __restrict__ pnorm, f16* __restrict__ pwn16) {
    long i = (long)blockIdx.x * blockDim.x + threadIdx.x;
    if (i >= (long)B_ * C_ * C_) return;
    int  o  = (int)(i % C_);
    long bc = i / C_;
    int  c  = (int)(bc % C_);
    int  b  = (int)(bc / C_);
    pwn16[i] = (f16)(p_w[i] / pnorm[b * C_ + o] * p_g[b * C_ + c]);
}

// ---------------------------------------------------------------------------
// GEMM1: h16[b,c,t] = f16(mish(sum_d w1[c,d]*x[b,t,d] + b1[c]))
//   A[m=c][k=d] = w1h (direct),  B[n=t][k=d] = x16 (direct)
// ---------------------------------------------------------------------------
__global__ __launch_bounds__(NTHREADS)
void k_gemm1_mish(const f16* __restrict__ w1h, const f16* __restrict__ xh,
                  const float* __restrict__ b1, f16* __restrict__ h16) {
    __shared__ __align__(16) f16 As[BM * STRIDE_H];
    __shared__ __align__(16) f16 Bs[BN * STRIDE_H];
    int tid = threadIdx.x, lane = tid & 31, wave = tid >> 5;
    int wm = wave >> 1, wn = wave & 1;
    int l15 = lane & 15, lhi = lane >> 4;
    int b  = blockIdx.z;
    int m0 = blockIdx.y * BM;   // c
    int n0 = blockIdx.x * BN;   // t
    const f16* Abase = w1h + (long)m0 * D_;
    const f16* Bbase = xh + ((long)b * T_ + n0) * D_;

    v8f acc[2][4] = {};
    for (int k0 = 0; k0 < D_; k0 += BK) {
        stage_direct(As, Abase + k0, D_, tid);
        stage_direct(Bs, Bbase + k0, D_, tid);
        if (k0 + BK < D_) {
            __builtin_prefetch(Abase + (long)(tid >> 1) * D_ + k0 + BK, 0, 1);
            __builtin_prefetch(Bbase + (long)(tid >> 1) * D_ + k0 + BK, 0, 1);
        }
        stage_wait_barrier();
        v16h a[2], bf[4];
#pragma unroll
        for (int mt = 0; mt < 2; ++mt) a[mt] = frag_a(As, wm * 32 + mt * 16 + l15, lhi);
#pragma unroll
        for (int nt = 0; nt < 4; ++nt) bf[nt] = frag_b(Bs, wn * 64 + nt * 16 + l15, lhi);
#pragma unroll
        for (int mt = 0; mt < 2; ++mt)
#pragma unroll
            for (int nt = 0; nt < 4; ++nt)
                acc[mt][nt] = __builtin_amdgcn_wmma_f32_16x16x32_f16(
                    false, a[mt], false, bf[nt], (short)0, acc[mt][nt], false, false);
        __syncthreads();
    }
#pragma unroll
    for (int mt = 0; mt < 2; ++mt)
#pragma unroll
        for (int nt = 0; nt < 4; ++nt)
#pragma unroll
            for (int v = 0; v < 8; ++v) {
                int m = m0 + wm * 32 + mt * 16 + 8 * lhi + v;   // c
                int n = n0 + wn * 64 + nt * 16 + l15;           // t
                float val = acc[mt][nt][v] + b1[m];
                h16[((long)b * C_ + m) * T_ + n] = (f16)mishf(val);
            }
}

// ---------------------------------------------------------------------------
// Depthwise conv K=9, zero pad 4:  conv16[b,c,t] = sum_k h[b,c,t+k-4]*w[k] + d_b
// ---------------------------------------------------------------------------
__global__ __launch_bounds__(256)
void k_dwconv(const f16* __restrict__ h16, const float* __restrict__ dwn,
              const float* __restrict__ d_b, f16* __restrict__ conv16) {
    int bc = blockIdx.x;                      // b*C + c
    __shared__ f16 sh[T_ + 2 * PAD_];
    const f16* hrow = h16 + (long)bc * T_;
    for (int t = threadIdx.x; t < T_; t += 256) sh[PAD_ + t] = hrow[t];
    if (threadIdx.x < PAD_) {
        sh[threadIdx.x] = (f16)0.f;
        sh[PAD_ + T_ + threadIdx.x] = (f16)0.f;
    }
    float w[K_];
#pragma unroll
    for (int k = 0; k < K_; ++k) w[k] = dwn[(long)bc * K_ + k];
    float bias = d_b[bc];
    __syncthreads();
    for (int t = threadIdx.x; t < T_; t += 256) {
        float a = bias;
#pragma unroll
        for (int k = 0; k < K_; ++k) a += (float)sh[t + k] * w[k];
        conv16[(long)bc * T_ + t] = (f16)a;
    }
}

// ---------------------------------------------------------------------------
// GEMM2: a2[b,o,t] = f16(mish(sum_c conv[b,c,t]*pwn[b,c,o] + p_b[b,o]))
//   A[m=o][k=c] = pwn16 (transposed stage), B[n=t][k=c] = conv16 (transposed)
// ---------------------------------------------------------------------------
__global__ __launch_bounds__(NTHREADS)
void k_gemm2_mish(const f16* __restrict__ pwn16, const f16* __restrict__ conv16,
                  const float* __restrict__ p_b, f16* __restrict__ act216) {
    __shared__ __align__(16) f16 As[BM * STRIDE_H];
    __shared__ __align__(16) f16 Bs[BN * STRIDE_H];
    int tid = threadIdx.x, lane = tid & 31, wave = tid >> 5;
    int wm = wave >> 1, wn = wave & 1;
    int l15 = lane & 15, lhi = lane >> 4;
    int b  = blockIdx.z;
    int m0 = blockIdx.y * BM;   // o
    int n0 = blockIdx.x * BN;   // t

    v8f acc[2][4] = {};
    for (int k0 = 0; k0 < C_; k0 += BK) {
        stage_transposed(As, pwn16 + ((long)b * C_ + k0) * C_ + m0, C_, tid);
        stage_transposed(Bs, conv16 + ((long)b * C_ + k0) * T_ + n0, T_, tid);
        if (k0 + BK < C_) {
            __builtin_prefetch(pwn16 + ((long)b * C_ + k0 + BK) * C_ + m0 + (tid & 127), 0, 1);
            __builtin_prefetch(conv16 + ((long)b * C_ + k0 + BK) * T_ + n0 + (tid & 127), 0, 1);
        }
        __syncthreads();
        v16h a[2], bf[4];
#pragma unroll
        for (int mt = 0; mt < 2; ++mt) a[mt] = frag_a(As, wm * 32 + mt * 16 + l15, lhi);
#pragma unroll
        for (int nt = 0; nt < 4; ++nt) bf[nt] = frag_b(Bs, wn * 64 + nt * 16 + l15, lhi);
#pragma unroll
        for (int mt = 0; mt < 2; ++mt)
#pragma unroll
            for (int nt = 0; nt < 4; ++nt)
                acc[mt][nt] = __builtin_amdgcn_wmma_f32_16x16x32_f16(
                    false, a[mt], false, bf[nt], (short)0, acc[mt][nt], false, false);
        __syncthreads();
    }
#pragma unroll
    for (int mt = 0; mt < 2; ++mt)
#pragma unroll
        for (int nt = 0; nt < 4; ++nt)
#pragma unroll
            for (int v = 0; v < 8; ++v) {
                int m = m0 + wm * 32 + mt * 16 + 8 * lhi + v;   // o
                int n = n0 + wn * 64 + nt * 16 + l15;           // t
                float val = acc[mt][nt][v] + p_b[b * C_ + m];
                act216[((long)b * C_ + m) * T_ + n] = (f16)mishf(val);
            }
}

// ---------------------------------------------------------------------------
// GEMM3: out[b,t,d] = sum_c w2[d,c]*a2[b,c,t] + b2[d] + x[b,t,d]
//   M=t (transposed stage of a2), N=d, B[n=d][k=c] = w2h (direct)
//   => epilogue stores f32 coalesced along d, fusing bias+residual+transpose.
// ---------------------------------------------------------------------------
__global__ __launch_bounds__(NTHREADS)
void k_gemm3_out(const f16* __restrict__ act216, const f16* __restrict__ w2h,
                 const float* __restrict__ b2, const float* __restrict__ x,
                 float* __restrict__ out) {
    __shared__ __align__(16) f16 As[BM * STRIDE_H];
    __shared__ __align__(16) f16 Bs[BN * STRIDE_H];
    int tid = threadIdx.x, lane = tid & 31, wave = tid >> 5;
    int wm = wave >> 1, wn = wave & 1;
    int l15 = lane & 15, lhi = lane >> 4;
    int b  = blockIdx.z;
    int m0 = blockIdx.y * BM;   // t
    int n0 = blockIdx.x * BN;   // d

    v8f acc[2][4] = {};
    for (int k0 = 0; k0 < C_; k0 += BK) {
        stage_transposed(As, act216 + ((long)b * C_ + k0) * T_ + m0, T_, tid);
        stage_direct(Bs, w2h + (long)n0 * C_ + k0, C_, tid);
        stage_wait_barrier();
        v16h a[2], bf[4];
#pragma unroll
        for (int mt = 0; mt < 2; ++mt) a[mt] = frag_a(As, wm * 32 + mt * 16 + l15, lhi);
#pragma unroll
        for (int nt = 0; nt < 4; ++nt) bf[nt] = frag_b(Bs, wn * 64 + nt * 16 + l15, lhi);
#pragma unroll
        for (int mt = 0; mt < 2; ++mt)
#pragma unroll
            for (int nt = 0; nt < 4; ++nt)
                acc[mt][nt] = __builtin_amdgcn_wmma_f32_16x16x32_f16(
                    false, a[mt], false, bf[nt], (short)0, acc[mt][nt], false, false);
        __syncthreads();
    }
#pragma unroll
    for (int mt = 0; mt < 2; ++mt)
#pragma unroll
        for (int nt = 0; nt < 4; ++nt)
#pragma unroll
            for (int v = 0; v < 8; ++v) {
                int m = m0 + wm * 32 + mt * 16 + 8 * lhi + v;   // t
                int n = n0 + wn * 64 + nt * 16 + l15;           // d
                long oidx = ((long)b * T_ + m) * D_ + n;
                out[oidx] = acc[mt][nt][v] + b2[n] + x[oidx];
            }
}

// ---------------------------------------------------------------------------
extern "C" void kernel_launch(void* const* d_in, const int* in_sizes, int n_in,
                              void* d_out, int out_size, void* d_ws, size_t ws_size,
                              hipStream_t stream) {
    const float* x   = (const float*)d_in[0];
    const float* d_w = (const float*)d_in[1];
    const float* d_g = (const float*)d_in[2];
    const float* d_b = (const float*)d_in[3];
    const float* p_w = (const float*)d_in[4];
    const float* p_g = (const float*)d_in[5];
    const float* p_b = (const float*)d_in[6];
    const float* w1  = (const float*)d_in[7];
    const float* b1  = (const float*)d_in[8];
    const float* w2  = (const float*)d_in[9];
    const float* b2  = (const float*)d_in[10];
    float* out = (float*)d_out;

    char* ws = (char*)d_ws;
    auto alloc = [&](size_t bytes) -> char* {
        char* p = ws;
        ws += (bytes + 255) & ~(size_t)255;
        return p;
    };
    f16* x16    = (f16*)alloc(sizeof(f16) * (size_t)B_ * T_ * D_);
    f16* w1h    = (f16*)alloc(sizeof(f16) * (size_t)C_ * D_);
    f16* w2h    = (f16*)alloc(sizeof(f16) * (size_t)D_ * C_);
    f16* h16    = (f16*)alloc(sizeof(f16) * (size_t)B_ * C_ * T_);
    f16* conv16 = (f16*)alloc(sizeof(f16) * (size_t)B_ * C_ * T_);
    f16* act216 = (f16*)alloc(sizeof(f16) * (size_t)B_ * C_ * T_);
    f16* pwn16  = (f16*)alloc(sizeof(f16) * (size_t)B_ * C_ * C_);
    float* dwn  = (float*)alloc(sizeof(float) * (size_t)B_ * C_ * K_);
    float* pnm  = (float*)alloc(sizeof(float) * (size_t)B_ * C_);

    // Pack f32 -> f16
    {
        int n = B_ * T_ * D_;
        k_f32_to_f16<<<(n + 255) / 256, 256, 0, stream>>>(x, x16, n);
    }
    {
        int n = C_ * D_;
        k_f32_to_f16<<<(n + 255) / 256, 256, 0, stream>>>(w1, w1h, n);
        k_f32_to_f16<<<(n + 255) / 256, 256, 0, stream>>>(w2, w2h, n);
    }

    // Weight norms
    k_dnorm<<<B_ * K_, NTHREADS, 0, stream>>>(d_w, d_g, dwn);
    k_pnorm<<<dim3(C_ / 256, B_), 256, 0, stream>>>(p_w, pnm);
    {
        long n = (long)B_ * C_ * C_;
        k_pwn<<<(unsigned)((n + 255) / 256), 256, 0, stream>>>(p_w, p_g, pnm, pwn16);
    }

    // Pipeline
    k_gemm1_mish<<<dim3(T_ / BN, C_ / BM, B_), NTHREADS, 0, stream>>>(w1h, x16, b1, h16);
    k_dwconv<<<B_ * C_, 256, 0, stream>>>(h16, dwn, d_b, conv16);
    k_gemm2_mish<<<dim3(T_ / BN, C_ / BM, B_), NTHREADS, 0, stream>>>(pwn16, conv16, p_b, act216);
    k_gemm3_out<<<dim3(D_ / BN, T_ / BM, B_), NTHREADS, 0, stream>>>(act216, w2h, b2, x, out);
}
